// BertSelfAttention_43404939493966
// MI455X (gfx1250) — matrix operand order repaired and checked by reading, not compiled, not run on hardware
//
#include <hip/hip_runtime.h>

#define SEQ    1024
#define HID    1024
#define NHEADS 16
#define HDIM   64
#define BATCH  4
#define KPAD   72   // padded LDS row stride (halfs) to spread banks

typedef __attribute__((ext_vector_type(16))) __bf16         v16bf;
typedef __attribute__((ext_vector_type(8)))  __bf16         v8bf;
typedef __attribute__((ext_vector_type(8)))  float          v8f;
typedef __attribute__((ext_vector_type(8)))  unsigned short u16x8;
typedef __attribute__((ext_vector_type(4)))  unsigned short u16x4;
typedef __attribute__((ext_vector_type(4)))  int            v4i;

#if __has_builtin(__builtin_amdgcn_global_load_async_to_lds_b128) && \
    __has_builtin(__builtin_amdgcn_s_wait_asynccnt)
#define USE_ASYNC_LDS 1
typedef __attribute__((address_space(1))) v4i gv4i;   // global int4
typedef __attribute__((address_space(3))) v4i lv4i;   // LDS int4
#else
#define USE_ASYNC_LDS 0
#endif

static __device__ __forceinline__ unsigned short f2bf(float f) {
  union { float f; unsigned int u; } x; x.f = f;
  unsigned int r = x.u + 0x7FFFu + ((x.u >> 16) & 1u);  // round-to-nearest-even
  return (unsigned short)(r >> 16);
}

static __device__ __forceinline__ v16bf cat16(v8bf lo, v8bf hi) {
  return __builtin_shufflevector(lo, hi, 0,1,2,3,4,5,6,7,8,9,10,11,12,13,14,15);
}

// 16-bit A fragment (16xK row-striped): lanes 0-15 hold K=k0+0..7 & k0+16..23,
// lanes 16-31 hold K=k0+8..15 & k0+24..31. 'row' already points at this lane's M row.
static __device__ __forceinline__ v16bf load_a_frag(const unsigned short* row, int k0, int half) {
  const unsigned short* p = row + k0 + half * 8;
  v8bf lo = *(const v8bf*)(p);
  v8bf hi = *(const v8bf*)(p + 16);
  return cat16(lo, hi);
}

// 16-bit B fragment (Kx16): lanes 0-15 hold K=k0+0..7 & k0+16..23,
// lanes 16-31 K=k0+8..15 & k0+24..31. 'colrow' points at this lane's N column (K-contiguous).
static __device__ __forceinline__ v16bf load_b_frag(const unsigned short* colrow, int k0, int half) {
  const unsigned short* p = colrow + k0 + half * 8;
  v8bf lo = *(const v8bf*)(p);
  v8bf hi = *(const v8bf*)(p + 16);
  return cat16(lo, hi);
}

// ---------------------------------------------------------------- cast f32 -> bf16
__global__ void cast_bf16_kernel(const float* __restrict__ in,
                                 unsigned short* __restrict__ out, int n4) {
  int i = blockIdx.x * blockDim.x + threadIdx.x;
  if (i < n4) {
    float4 v = ((const float4*)in)[i];
    u16x4 o;
    o[0] = f2bf(v.x); o[1] = f2bf(v.y); o[2] = f2bf(v.z); o[3] = f2bf(v.w);
    ((u16x4*)out)[i] = o;
  }
}

// ---------------------------------------------------------------- QKV projection GEMM
// C[m,n] = sum_k A[m,k] * W[n,k] + bias[n]   (A: 4096x1024 bf16, W: 1024x1024 bf16)
// Block: 8 waves, each wave computes a 32x64 tile -> block covers 256x64 of C.
// Output stored head-split bf16: Out[b][h][s][d],  m = b*SEQ+s, n = h*HDIM+d.
__global__ void __launch_bounds__(256)
qkv_gemm_kernel(const unsigned short* __restrict__ Abf,
                const unsigned short* __restrict__ Wbf,
                const float* __restrict__ bias,
                unsigned short* __restrict__ Out) {
  const int lane = threadIdx.x & 31;
  const int wave = threadIdx.x >> 5;
  const int half = lane >> 4;
  const int l16  = lane & 15;

  const int m_base = blockIdx.y * 256 + wave * 32;
  const int n_base = blockIdx.x * 64;

  v8f acc[2][4] = {};
  const unsigned short* arow0 = Abf + (size_t)(m_base + l16) * HID;
  const unsigned short* arow1 = Abf + (size_t)(m_base + 16 + l16) * HID;

  for (int k0 = 0; k0 < HID; k0 += 32) {
    v16bf af0 = load_a_frag(arow0, k0, half);
    v16bf af1 = load_a_frag(arow1, k0, half);
#pragma unroll
    for (int t = 0; t < 4; ++t) {
      const unsigned short* wrow = Wbf + (size_t)(n_base + t * 16 + l16) * HID;
      v16bf bfrag = load_b_frag(wrow, k0, half);
      acc[0][t] = __builtin_amdgcn_wmma_f32_16x16x32_bf16(
          false, af0, false, bfrag, (short)0, acc[0][t], false, false);
      acc[1][t] = __builtin_amdgcn_wmma_f32_16x16x32_bf16(
          false, af1, false, bfrag, (short)0, acc[1][t], false, false);
    }
  }

#pragma unroll
  for (int t = 0; t < 4; ++t) {
    int n = n_base + t * 16 + l16;
    float bv = bias[n];
    int h = n >> 6, d = n & (HDIM - 1);
#pragma unroll
    for (int sub = 0; sub < 2; ++sub) {
#pragma unroll
      for (int r = 0; r < 8; ++r) {
        int m = m_base + sub * 16 + r + half * 8;
        int b = m >> 10, s = m & (SEQ - 1);
        size_t off = (((size_t)(b * NHEADS + h) * SEQ) + s) * HDIM + d;
        Out[off] = f2bf(acc[sub][t][r] + bv);
      }
    }
  }
}

// ---------------------------------------------------------------- fused flash attention
// Grid: (SEQ/64, NHEADS, BATCH), 128 threads (4 waves), each wave owns 16 q rows.
__global__ void __launch_bounds__(128)
attn_kernel(const unsigned short* __restrict__ Q,   // [B][NH][S][HD] bf16
            const unsigned short* __restrict__ K,
            const unsigned short* __restrict__ V,
            const float* __restrict__ mask,         // [B][1][S][S]
            const int* __restrict__ sep_idx,        // [B][3]
            const float* __restrict__ w0,
            const float* __restrict__ w1,
            float* __restrict__ out) {              // [B][S][H]
  __shared__ unsigned short Ksh[64][KPAD];          // K tile, row-major (kk, d)
  __shared__ unsigned short Vt[HDIM][KPAD];         // V tile transposed (d, kk)
  __shared__ unsigned short Psh[4][16][KPAD];       // per-wave P scratch

  const int lane = threadIdx.x & 31;
  const int wave = threadIdx.x >> 5;
  const int half = lane >> 4;
  const int l16  = lane & 15;

  const int b = blockIdx.z;
  const int h = blockIdx.y;
  const int q_base = blockIdx.x * 64 + wave * 16;

  const unsigned short* Qh = Q + (size_t)(b * NHEADS + h) * SEQ * HDIM;
  const unsigned short* Kh = K + (size_t)(b * NHEADS + h) * SEQ * HDIM;
  const unsigned short* Vh = V + (size_t)(b * NHEADS + h) * SEQ * HDIM;
  const float* maskb = mask + (size_t)b * SEQ * SEQ;

  const float w0c  = fminf(fmaxf(w0[0], 0.0f), 0.5f);
  const float w1c  = fminf(fmaxf(w1[0], 0.5f), 1.0f);
  const int   idx2 = sep_idx[b * 3 + 2];
  const float inv_sqrt = 0.125f;  // 1/sqrt(64)

  // Q strip for this wave, kept resident: two K-steps over HD=64
  const unsigned short* qrow = Qh + (size_t)(q_base + l16) * HDIM;
  v16bf qf[2];
#pragma unroll
  for (int ks = 0; ks < 2; ++ks) qf[ks] = load_a_frag(qrow, ks * 32, half);

  v8f ctx[4] = {};
  float mrow[8], lrow[8];
#pragma unroll
  for (int r = 0; r < 8; ++r) { mrow[r] = -3.0e38f; lrow[r] = 0.0f; }

  for (int kt = 0; kt < SEQ / 64; ++kt) {
    const int kk0 = kt * 64;
    __syncthreads();  // previous iteration's LDS reads done before overwrite

    // stage K tile (row-major) and V tile (transposed) into LDS
    for (int i = threadIdx.x; i < 64 * (HDIM / 8); i += blockDim.x) {
      int rr = i >> 3;
      int cc = (i & 7) * 8;
#if USE_ASYNC_LDS
      // CDNA5 async DMA path: global -> LDS directly, tracked by ASYNCcnt
      __builtin_amdgcn_global_load_async_to_lds_b128(
          (gv4i*)(Kh + (size_t)(kk0 + rr) * HDIM + cc),
          (lv4i*)(&Ksh[rr][cc]), 0, 0);
#else
      v8bf kd = *(const v8bf*)(Kh + (size_t)(kk0 + rr) * HDIM + cc);
      *(v8bf*)(&Ksh[rr][cc]) = kd;
#endif
      v8bf vd = *(const v8bf*)(Vh + (size_t)(kk0 + rr) * HDIM + cc);
      u16x8 vu = __builtin_bit_cast(u16x8, vd);
#pragma unroll
      for (int j = 0; j < 8; ++j) Vt[cc + j][rr] = vu[j];
    }
    if (kt + 1 < SEQ / 64) {  // pull next tile toward L2 (global_prefetch_b8)
      __builtin_prefetch(Kh + (size_t)(kk0 + 64 + (threadIdx.x >> 1)) * HDIM, 0, 1);
      __builtin_prefetch(Vh + (size_t)(kk0 + 64 + (threadIdx.x >> 1)) * HDIM, 0, 1);
    }
#if USE_ASYNC_LDS
    __builtin_amdgcn_s_wait_asynccnt(0);
#endif
    __syncthreads();

    // scores: S[16 x 64] = Q(16x64) x K_tile^T  via bf16 WMMA, f32 accum
    v8f sc[4] = {};
#pragma unroll
    for (int t = 0; t < 4; ++t) {
      const unsigned short* krow = &Ksh[t * 16 + l16][0];
#pragma unroll
      for (int ks = 0; ks < 2; ++ks) {
        v16bf kf = load_b_frag(krow, ks * 32, half);
        sc[t] = __builtin_amdgcn_wmma_f32_16x16x32_bf16(
            false, qf[ks], false, kf, (short)0, sc[t], false, false);
      }
    }

    // scale (1/sqrt(d)), segment weights, additive mask
#pragma unroll
    for (int t = 0; t < 4; ++t) {
      int kcol = kk0 + t * 16 + l16;
      bool col_ge = (kcol >= idx2);
#pragma unroll
      for (int r = 0; r < 8; ++r) {
        int qr = q_base + r + half * 8;
        float scl = col_ge ? ((qr < idx2) ? w0c : w1c) : 1.0f;
        sc[t][r] = sc[t][r] * inv_sqrt * scl + maskb[(size_t)qr * SEQ + kcol];
      }
    }

    // online softmax per row (row r lives in VGPR r across the 16-lane half)
#pragma unroll
    for (int r = 0; r < 8; ++r) {
      float tm = fmaxf(fmaxf(sc[0][r], sc[1][r]), fmaxf(sc[2][r], sc[3][r]));
      tm = fmaxf(tm, __shfl_xor(tm, 1));
      tm = fmaxf(tm, __shfl_xor(tm, 2));
      tm = fmaxf(tm, __shfl_xor(tm, 4));
      tm = fmaxf(tm, __shfl_xor(tm, 8));
      float mnew = fmaxf(mrow[r], tm);
      float corr = __expf(mrow[r] - mnew);
      float rsum = 0.0f;
#pragma unroll
      for (int t = 0; t < 4; ++t) {
        float p = __expf(sc[t][r] - mnew);
        sc[t][r] = p;
        rsum += p;
      }
      rsum += __shfl_xor(rsum, 1);
      rsum += __shfl_xor(rsum, 2);
      rsum += __shfl_xor(rsum, 4);
      rsum += __shfl_xor(rsum, 8);
      lrow[r] = lrow[r] * corr + rsum;
      mrow[r] = mnew;
#pragma unroll
      for (int t = 0; t < 4; ++t) ctx[t][r] *= corr;
    }

    // P (C-layout f32) -> bf16 A-fragment via per-wave LDS round trip
#pragma unroll
    for (int t = 0; t < 4; ++t)
#pragma unroll
      for (int r = 0; r < 8; ++r)
        Psh[wave][r + half * 8][t * 16 + l16] = f2bf(sc[t][r]);

    // ctx += P(16x64) x V_tile(64x64)
#pragma unroll
    for (int ks = 0; ks < 2; ++ks) {
      v16bf pf = load_a_frag(&Psh[wave][l16][0], ks * 32, half);
#pragma unroll
      for (int t = 0; t < 4; ++t) {
        v16bf vf = load_b_frag(&Vt[t * 16 + l16][0], ks * 32, half);
        ctx[t] = __builtin_amdgcn_wmma_f32_16x16x32_bf16(
            false, pf, false, vf, (short)0, ctx[t], false, false);
      }
    }
  }

  // finalize: divide by softmax denominator, merge heads, store f32
#pragma unroll
  for (int t = 0; t < 4; ++t) {
#pragma unroll
    for (int r = 0; r < 8; ++r) {
      int qr = q_base + r + half * 8;
      float o = ctx[t][r] / lrow[r];
      out[((size_t)b * SEQ + qr) * HID + h * HDIM + t * 16 + l16] = o;
    }
  }
}

// ---------------------------------------------------------------- launcher
extern "C" void kernel_launch(void* const* d_in, const int* in_sizes, int n_in,
                              void* d_out, int out_size, void* d_ws, size_t ws_size,
                              hipStream_t stream) {
  const float* hs   = (const float*)d_in[0];
  const float* mask = (const float*)d_in[1];
  const int*   sep  = (const int*)d_in[2];
  const float* Wq   = (const float*)d_in[3];
  const float* bq   = (const float*)d_in[4];
  const float* Wk   = (const float*)d_in[5];
  const float* bk   = (const float*)d_in[6];
  const float* Wv   = (const float*)d_in[7];
  const float* bv   = (const float*)d_in[8];
  const float* w0   = (const float*)d_in[9];
  const float* w1   = (const float*)d_in[10];
  float* out = (float*)d_out;

  const size_t MB = 1u << 20;
  char* w = (char*)d_ws;
  unsigned short* hid_bf = (unsigned short*)(w);             //  8 MB
  unsigned short* wq_bf  = (unsigned short*)(w + 8  * MB);   //  2 MB
  unsigned short* wk_bf  = (unsigned short*)(w + 10 * MB);   //  2 MB
  unsigned short* wv_bf  = (unsigned short*)(w + 12 * MB);   //  2 MB
  unsigned short* q_bf   = (unsigned short*)(w + 14 * MB);   //  8 MB
  unsigned short* k_bf   = (unsigned short*)(w + 22 * MB);   //  8 MB
  unsigned short* v_bf   = (unsigned short*)(w + 30 * MB);   //  8 MB

  // casts
  {
    int n4 = (BATCH * SEQ * HID) / 4;   // 1,048,576
    cast_bf16_kernel<<<(n4 + 255) / 256, 256, 0, stream>>>(hs, hid_bf, n4);
    int w4 = (HID * HID) / 4;           //   262,144
    cast_bf16_kernel<<<(w4 + 255) / 256, 256, 0, stream>>>(Wq, wq_bf, w4);
    cast_bf16_kernel<<<(w4 + 255) / 256, 256, 0, stream>>>(Wk, wk_bf, w4);
    cast_bf16_kernel<<<(w4 + 255) / 256, 256, 0, stream>>>(Wv, wv_bf, w4);
  }

  // QKV projections: grid = (N/64, M/256) = (16, 16), 256 threads
  dim3 ggrid(HID / 64, (BATCH * SEQ) / 256);
  qkv_gemm_kernel<<<ggrid, 256, 0, stream>>>(hid_bf, wq_bf, bq, q_bf);
  qkv_gemm_kernel<<<ggrid, 256, 0, stream>>>(hid_bf, wk_bf, bk, k_bf);
  qkv_gemm_kernel<<<ggrid, 256, 0, stream>>>(hid_bf, wv_bf, bv, v_bf);

  // fused attention: grid = (S/64, NH, B), 128 threads (4 waves)
  dim3 agrid(SEQ / 64, NHEADS, BATCH);
  attn_kernel<<<agrid, 128, 0, stream>>>(q_bf, k_bf, v_bf, mask, sep, w0, w1, out);
}
